// MultiHead_Attention_Lattice_rel_35029753266235
// MI455X (gfx1250) — matrix operand order repaired and checked by reading, not compiled, not run on hardware
//
#include <hip/hip_runtime.h>
#include <hip/hip_bf16.h>

// ---------------------------------------------------------------------------
// MultiHead_Attention_Lattice_rel for MI455X (gfx1250, wave32, WMMA)
//   B=4, S=384, H=256, NH=8, DH=32
// Pipeline:
//   1) proj_kernel       : q/k/v = X @ W + b                (bf16 out, WMMA)
//   2) qw_kernel         : qW[b,s,h,e] = sum_d q*Wr ; qbr   (VALU, tiny)
//   3) scores_kernel     : scores[b,h,q,k] = A_C + B_D + bias, masked (WMMA)
//                          -- streams the 604MB rel tensor exactly once,
//                             via async global->LDS DMA (ASYNCcnt), double
//                             buffered per wave
//   4) softmax_out_kernel: row softmax + attn @ V           (WMMA)
// ---------------------------------------------------------------------------

#define B_  4
#define S_  384
#define H_  256
#define NH_ 8
#define DH_ 32

#if defined(__gfx1250__) && __has_builtin(__builtin_amdgcn_global_load_async_to_lds_b128)
#define HAVE_ASYNC 1
#else
#define HAVE_ASYNC 0
#endif

typedef __attribute__((ext_vector_type(16))) __bf16 v16bf;
typedef __attribute__((ext_vector_type(8)))  float  v8f;
typedef __attribute__((ext_vector_type(4)))  int    v4i;

__device__ __forceinline__ v8f wmma_bf16(v16bf a, v16bf b, v8f c) {
  // D = A(16x32 bf16) * B(32x16 bf16) + C(16x16 f32)
  return __builtin_amdgcn_wmma_f32_16x16x32_bf16(
      /*neg_a=*/false, a, /*neg_b=*/false, b,
      /*c_mod=*/(short)0, c, /*reuse_a=*/false, /*reuse_b=*/false);
}

// A-fragment element->K mapping (16-bit A 16x32, wave32):
//   lanes 0-15 : K = 0..7 (elems 0-7), 16..23 (elems 8-15)
//   lanes 16-31: K = 8..15,            24..31
__device__ __forceinline__ int amap(int j, int lane) {
  return ((lane & 16) ? 8 : 0) + (j < 8 ? j : j + 8);
}

#if HAVE_ASYNC
// Issue one 16(row) x 32(e) f32 chunk of rel into this wave's LDS buffer.
// 512 dwords -> 4 async b128 instructions per wave (one per lane each iter).
__device__ __forceinline__ void async_rel_chunk(const float* __restrict__ gsrc,
                                                float* lds_base, int lane) {
  const int row = lane >> 3;        // 0..3
  const int ep  = (lane & 7) * 4;   // 0,4,...,28
#pragma unroll
  for (int i = 0; i < 4; ++i) {
    const float* g = gsrc + (size_t)(i * 4 + row) * H_ + ep;
    float* l = lds_base + (i * 4 + row) * 32 + ep;
    __builtin_amdgcn_global_load_async_to_lds_b128(
        (__attribute__((address_space(1))) v4i*)g,
        (__attribute__((address_space(3))) v4i*)l,
        /*offset=*/0, /*cpol=*/0);
  }
}

__device__ __forceinline__ void wait_async_le(int pending) {
#if __has_builtin(__builtin_amdgcn_s_wait_asynccnt)
  if (pending) __builtin_amdgcn_s_wait_asynccnt(4);
  else         __builtin_amdgcn_s_wait_asynccnt(0);
#else
  if (pending) asm volatile("s_wait_asynccnt 4" ::: "memory");
  else         asm volatile("s_wait_asynccnt 0" ::: "memory");
#endif
  asm volatile("" ::: "memory");
}
#endif  // HAVE_ASYNC

// ---------------------------------------------------------------------------
// Kernel 1: OUT[r, n] = bf16( X[r,:] @ W[:,n] + bias[n] ),  X:[B*S,H] W:[H,H]
// grid.x = (B*S)/16 ; block = 256 (8 waves). Wave w owns N-tiles 2w, 2w+1.
// ---------------------------------------------------------------------------
__global__ void proj_kernel(const float* __restrict__ X,
                            const float* __restrict__ W,
                            const float* __restrict__ bias,
                            __bf16* __restrict__ OUT) {
  const int w    = threadIdx.x >> 5;
  const int lane = threadIdx.x & 31;
  const int row0 = blockIdx.x * 16;
  const int m    = lane & 15;
  const int nt0  = w * 2, nt1 = w * 2 + 1;

  v8f c0 = {}, c1 = {};
#pragma unroll
  for (int kc = 0; kc < 8; ++kc) {
    v16bf a;
#pragma unroll
    for (int j = 0; j < 16; ++j) {
      int e = kc * 32 + amap(j, lane);
      a[j] = (__bf16)X[(size_t)(row0 + m) * H_ + e];
    }
    v16bf b0, b1;
#pragma unroll
    for (int j = 0; j < 16; ++j) {
      const float* wp = W + (size_t)(kc * 32 + amap(j, lane)) * H_;
      b0[j] = (__bf16)wp[nt0 * 16 + (lane & 15)];
      b1[j] = (__bf16)wp[nt1 * 16 + (lane & 15)];
    }
    c0 = wmma_bf16(a, b0, c0);
    c1 = wmma_bf16(a, b1, c1);
  }
  const int n = lane & 15;
  const int rb = (lane >> 4) * 8;
#pragma unroll
  for (int r = 0; r < 8; ++r) {
    int row = row0 + rb + r;
    OUT[(size_t)row * H_ + nt0 * 16 + n] = (__bf16)(c0[r] + bias[nt0 * 16 + n]);
    OUT[(size_t)row * H_ + nt1 * 16 + n] = (__bf16)(c1[r] + bias[nt1 * 16 + n]);
  }
}

// ---------------------------------------------------------------------------
// Kernel 2: qW[bs,h,e] = sum_d q[bs, h*32+d] * Wr[e, h*32+d]   (bf16 out)
//           qbr[bs,h]  = sum_d q[bs, h*32+d] * br[h*32+d]      (f32)
// grid = (B*S, NH), block = 256 (thread = e)
// ---------------------------------------------------------------------------
__global__ void qw_kernel(const __bf16* __restrict__ qb,
                          const float* __restrict__ Wr,
                          const float* __restrict__ br,
                          __bf16* __restrict__ qWb,
                          float* __restrict__ qbr) {
  const int bs = blockIdx.x;
  const int h  = blockIdx.y;
  const int e  = threadIdx.x;

  float qv[32];
#pragma unroll
  for (int d = 0; d < 32; ++d) qv[d] = (float)qb[(size_t)bs * H_ + h * 32 + d];

  const float* wrp = Wr + (size_t)e * H_ + h * 32;
  float acc = 0.f;
#pragma unroll
  for (int d = 0; d < 32; ++d) acc += qv[d] * wrp[d];
  qWb[((size_t)bs * NH_ + h) * H_ + e] = (__bf16)acc;

  if (e == 0) {
    float a2 = 0.f;
#pragma unroll
    for (int d = 0; d < 32; ++d) a2 += qv[d] * br[h * 32 + d];
    qbr[(size_t)bs * NH_ + h] = a2;
  }
}

// ---------------------------------------------------------------------------
// Kernel 3: scores[b,h,q,k] = q.k (content) + q.rel (lattice) + q.br, masked.
// One wave per query row. A-matrix rows = heads (8 used + 8 pad):
//   A_C : A[h,e] = q[b,q,e] if e in head-h block else 0 ; B[e,k] = k_proj
//   B_D : A[h,e] = qW[b,q,h,e]                          ; B[e,k] = rel_raw
// Both accumulate into the same C tile -> 16 WMMAs per (qrow, ktile).
// rel_raw streams HBM->LDS via async DMA, double-buffered per wave (private
// LDS slots, no workgroup barriers; ordering via ASYNCcnt only).
// grid = (S/8, B), block = 256 (8 waves).
// ---------------------------------------------------------------------------
__global__ void scores_kernel(const __bf16* __restrict__ qb,
                              const __bf16* __restrict__ kb,
                              const __bf16* __restrict__ qWb,
                              const float* __restrict__ qbr,
                              const float* __restrict__ rel,
                              const int* __restrict__ seq_len,
                              const int* __restrict__ lex_num,
                              float* __restrict__ scores) {
  const int w    = threadIdx.x >> 5;
  const int lane = threadIdx.x & 31;
  const int qrow = blockIdx.x * 8 + w;
  const int b    = blockIdx.y;
  const int hrow = lane & 15;

#if HAVE_ASYNC
  // per-wave double buffer: 2 x (16 rows x 32 e) f32 = 4KB ; 8 waves = 32KB
  __shared__ float relbuf[8][2][512];
#endif

  // Build the two A operands for the full K=256 once per q row.
  v16bf aQ[8], aW[8];
#pragma unroll
  for (int ec = 0; ec < 8; ++ec) {
    v16bf aq, aw;
#pragma unroll
    for (int j = 0; j < 16; ++j) {
      int e = ec * 32 + amap(j, lane);
      __bf16 qv = (__bf16)0.0f, wv = (__bf16)0.0f;
      if (hrow < 8) {
        if ((e >> 5) == hrow) qv = qb[((size_t)b * S_ + qrow) * H_ + e];
        wv = qWb[(((size_t)b * S_ + qrow) * NH_ + hrow) * H_ + e];
      }
      aq[j] = qv;
      aw[j] = wv;
    }
    aQ[ec] = aq;
    aW[ec] = aw;
  }

  float hb[8];
#pragma unroll
  for (int r = 0; r < 8; ++r) hb[r] = qbr[((size_t)b * S_ + qrow) * NH_ + r];

  const int limit = seq_len[b] + lex_num[b];
  const float* relrow = rel + ((size_t)b * S_ + qrow) * (size_t)S_ * H_;

#if HAVE_ASYNC
  // prologue: chunk (kt=0, ec=0) -> buffer 0   (buffer parity == ec & 1)
  async_rel_chunk(relrow, relbuf[w][0], lane);
#endif

  for (int kt = 0; kt < S_ / 16; ++kt) {
    v8f c = {};
#pragma unroll
    for (int ec = 0; ec < 8; ++ec) {
      v16bf brl;
#if HAVE_ASYNC
      {
        const int nkt = (ec == 7) ? kt + 1 : kt;
        const int nec = (ec + 1) & 7;
        const bool more = (nkt < S_ / 16);
        if (more)
          async_rel_chunk(relrow + (size_t)(nkt * 16) * H_ + nec * 32,
                          relbuf[w][(ec + 1) & 1], lane);
        wait_async_le(more ? 1 : 0);
        const float* buf = relbuf[w][ec & 1];
#pragma unroll
        for (int j = 0; j < 16; ++j)         // lane&15 == k-row, elem j == e (amap)
          brl[j] = (__bf16)buf[(lane & 15) * 32 + amap(j, lane)];
      }
#else
#pragma unroll
      for (int j = 0; j < 16; ++j)
        brl[j] = (__bf16)relrow[(size_t)(kt * 16 + (lane & 15)) * H_ + ec * 32 + amap(j, lane)];
#endif
      c = wmma_bf16(aW[ec], brl, c);         // lattice term

      v16bf bkk;                             // content term: B[e,k] = k_proj
#pragma unroll
      for (int j = 0; j < 16; ++j)
        bkk[j] = kb[((size_t)b * S_ + kt * 16 + (lane & 15)) * H_ + ec * 32 + amap(j, lane)];
      c = wmma_bf16(aQ[ec], bkk, c);
    }

    // epilogue: lanes 0-15 hold heads 0-7 in c[0..7]; col = lane
    if (lane < 16) {
      const int kcol = kt * 16 + lane;
      const bool ok  = kcol < limit;
#pragma unroll
      for (int r = 0; r < 8; ++r) {
        float sv = ok ? (c[r] + hb[r]) : -1e15f;
        scores[(((size_t)b * NH_ + r) * S_ + qrow) * S_ + kcol] = sv;
      }
    }
  }
}

// ---------------------------------------------------------------------------
// Kernel 4: row softmax over k + out[b,q,h*32+d] = attn @ V_head.
// One wave per (b, h, 16-query tile). Scores are L2-resident (18.9 MB).
// grid = B*NH*(S/16)/8 , block = 256 (8 waves).
// ---------------------------------------------------------------------------
__global__ void softmax_out_kernel(const float* __restrict__ scores,
                                   const __bf16* __restrict__ vb,
                                   float* __restrict__ out) {
  const int lane = threadIdx.x & 31;
  const int task = blockIdx.x * 8 + (threadIdx.x >> 5);
  const int QT   = S_ / 16;
  const int b    = task / (NH_ * QT);
  const int rem  = task % (NH_ * QT);
  const int h    = rem / QT;
  const int q0   = (rem % QT) * 16;

  const float* srow = scores + (((size_t)b * NH_ + h) * S_ + q0) * (size_t)S_;

  // Per-row max and 1/sum(exp). All 32 lanes cooperate on one row at a time;
  // lane keeps stats for the row it represents in the A-fragment (lane&15).
  float myMax = 0.f, myInv = 0.f;
  for (int r = 0; r < 16; ++r) {
    float mx = -3.0e38f;
    for (int i = 0; i < S_ / 32; ++i)
      mx = fmaxf(mx, srow[(size_t)r * S_ + i * 32 + lane]);
#pragma unroll
    for (int off = 16; off; off >>= 1) mx = fmaxf(mx, __shfl_xor(mx, off, 32));
    float sm = 0.f;
    for (int i = 0; i < S_ / 32; ++i)
      sm += __expf(srow[(size_t)r * S_ + i * 32 + lane] - mx);
#pragma unroll
    for (int off = 16; off; off >>= 1) sm += __shfl_xor(sm, off, 32);
    if ((lane & 15) == r) { myMax = mx; myInv = 1.0f / sm; }
  }

  // out tile [16 q, 32 d] = P(16xS) @ V_head(Sx32), K-chunked by 32.
  v8f c0 = {}, c1 = {};
  const int m = lane & 15;
  for (int kc = 0; kc < S_ / 32; ++kc) {
    v16bf pa;
#pragma unroll
    for (int j = 0; j < 16; ++j) {
      int k = kc * 32 + amap(j, lane);
      pa[j] = (__bf16)(__expf(srow[(size_t)m * S_ + k] - myMax) * myInv);
    }
    v16bf b0, b1;
#pragma unroll
    for (int j = 0; j < 16; ++j) {
      const __bf16* vbase = vb + ((size_t)b * S_ + kc * 32 + amap(j, lane)) * H_ + h * 32;
      b0[j] = vbase[(lane & 15)];
      b1[j] = vbase[16 + (lane & 15)];
    }
    c0 = wmma_bf16(pa, b0, c0);
    c1 = wmma_bf16(pa, b1, c1);
  }

  const int n  = lane & 15;
  const int rb = (lane >> 4) * 8;
#pragma unroll
  for (int r = 0; r < 8; ++r) {
    int q = q0 + rb + r;
    out[((size_t)b * S_ + q) * H_ + h * 32 + n]      = c0[r];
    out[((size_t)b * S_ + q) * H_ + h * 32 + 16 + n] = c1[r];
  }
}

// ---------------------------------------------------------------------------
extern "C" void kernel_launch(void* const* d_in, const int* in_sizes, int n_in,
                              void* d_out, int out_size, void* d_ws, size_t ws_size,
                              hipStream_t stream) {
  const float* key   = (const float*)d_in[0];
  const float* query = (const float*)d_in[1];
  const float* value = (const float*)d_in[2];
  const float* rel   = (const float*)d_in[3];
  const int*   seqlen = (const int*)d_in[4];
  const int*   lexnum = (const int*)d_in[5];
  const float* Wk = (const float*)d_in[6];
  const float* bk = (const float*)d_in[7];
  const float* Wq = (const float*)d_in[8];
  const float* bq = (const float*)d_in[9];
  const float* Wv = (const float*)d_in[10];
  const float* bv = (const float*)d_in[11];
  const float* Wr = (const float*)d_in[12];
  const float* br = (const float*)d_in[13];
  float* out = (float*)d_out;

  char* ws = (char*)d_ws;
  size_t off = 0;
  auto take = [&](size_t bytes) {
    char* p = ws + off;
    off = (off + bytes + 255) & ~(size_t)255;
    return p;
  };
  __bf16* qb   = (__bf16*)take((size_t)B_ * S_ * H_ * 2);
  __bf16* kb   = (__bf16*)take((size_t)B_ * S_ * H_ * 2);
  __bf16* vb   = (__bf16*)take((size_t)B_ * S_ * H_ * 2);
  __bf16* qWb  = (__bf16*)take((size_t)B_ * S_ * NH_ * H_ * 2);
  float*  qbr  = (float*)take((size_t)B_ * S_ * NH_ * 4);
  float*  scor = (float*)take((size_t)B_ * NH_ * S_ * S_ * 4);

  // 1) projections (bf16 outputs)
  proj_kernel<<<dim3((B_ * S_) / 16), 256, 0, stream>>>(query, Wq, bq, qb);
  proj_kernel<<<dim3((B_ * S_) / 16), 256, 0, stream>>>(key,   Wk, bk, kb);
  proj_kernel<<<dim3((B_ * S_) / 16), 256, 0, stream>>>(value, Wv, bv, vb);

  // 2) fold Wr into the query side: qW = q . Wr^T(head-block), qbr = q . br
  qw_kernel<<<dim3(B_ * S_, NH_), 256, 0, stream>>>(qb, Wr, br, qWb, qbr);

  // 3) fused content + lattice scores, masked (streams rel once, async->LDS)
  scores_kernel<<<dim3(S_ / 8, B_), 256, 0, stream>>>(qb, kb, qWb, qbr, rel,
                                                      seqlen, lexnum, scor);

  // 4) softmax + attn @ V  (scores stay in the 192MB L2)
  softmax_out_kernel<<<dim3((B_ * NH_ * (S_ / 16)) / 8), 256, 0, stream>>>(scor, vb, out);
}
